// VectorQuantizer_4191888081300
// MI455X (gfx1250) — matrix operand order repaired and checked by reading, not compiled, not run on hardware
//
#include <hip/hip_runtime.h>

// ---------------------------------------------------------------------------
// VQ-VAE vector quantizer for gfx1250 (MI455X).
// Fused [32768x256]x[256x8192] score GEMM + row argmin on
// v_wmma_f32_16x16x32_bf16 with split-bf16 (hi+lo) operands (~fp32 accuracy).
// Code tiles are staged in LDS (double-buffered) and shared by all 8 waves;
// each wave keeps 32 tokens of A resident in VGPRs for the whole sweep.
// ---------------------------------------------------------------------------

typedef __attribute__((ext_vector_type(16))) __bf16 v16bf;
typedef __attribute__((ext_vector_type(8)))  float  v8f;

#define TOKENS 32768      // B*H*W
#define CODES  8192
#define CH     256
#define NC     8388608    // TOKENS*CH

#define ROWB   528        // padded LDS row stride (256*2B + 16B) -> no bank conflicts
#define MATB   (16 * ROWB)        // one 16-code bf16 tile in LDS (8448 B)
#define BUFB   (2 * MATB)         // hi+lo tile (16896 B)
#define NT     256        // code tiles per grid-Y half (256*16 = 4096 codes)

union AFrag { v16bf v; uint4 q[2]; };

#define WMMA_BF16(A, B, C) \
  __builtin_amdgcn_wmma_f32_16x16x32_bf16(false, (A), false, (B), (short)0, (C), false, false)

// ---------------------------------------------------------------------------
// Split z (NCHW f32) into token-major bf16 hi/lo planes, pre-scaled by -2 so
// the WMMA accumulates  ||e||^2 - 2 z.e  directly.
// ---------------------------------------------------------------------------
__global__ void prep_z(const float* __restrict__ z,
                       __bf16* __restrict__ zh, __bf16* __restrict__ zl) {
  const int n = blockIdx.x;
  const int c = threadIdx.x;
  const int b = n >> 10;
  const int hw = n & 1023;
  float x = -2.0f * z[((size_t)(b * CH + c) << 10) + hw];
  __bf16 h = (__bf16)x;
  __bf16 l = (__bf16)(x - (float)h);
  zh[(size_t)n * CH + c] = h;
  zl[(size_t)n * CH + c] = l;
}

// ---------------------------------------------------------------------------
// Split embedding rows into bf16 hi/lo and compute ||e_k||^2.
// ---------------------------------------------------------------------------
__global__ void prep_e(const float* __restrict__ e,
                       __bf16* __restrict__ eh, __bf16* __restrict__ el,
                       float* __restrict__ en2) {
  __shared__ float red[256];
  const int k = blockIdx.x;
  const int c = threadIdx.x;
  float x = e[(size_t)k * CH + c];
  __bf16 h = (__bf16)x;
  __bf16 l = (__bf16)(x - (float)h);
  eh[(size_t)k * CH + c] = h;
  el[(size_t)k * CH + c] = l;
  red[c] = x * x;
  __syncthreads();
  for (int s = 128; s > 0; s >>= 1) {
    if (c < s) red[c] += red[c + s];
    __syncthreads();
  }
  if (c == 0) en2[k] = red[0];
}

// ---------------------------------------------------------------------------
// Fused GEMM + argmin.
//   grid = (128, 2): x -> 256-token block (8 waves x 32 tokens),
//                    y -> code half [0,4096) / [4096,8192)
// Per iteration one 16-code tile (hi+lo, 16 KB) is staged in LDS by all 256
// threads (double-buffered), then each wave runs 48 WMMAs against its two
// resident 16-token A strips, folding ||e||^2 in via the accumulator init.
// The ||e||^2 lane value is double-buffered across iterations so the acc init
// never waits on a load issued in the same iteration (keeps the next-tile
// global prefetch fully overlapped with the WMMA chain).
// ---------------------------------------------------------------------------
__global__ void __launch_bounds__(256) vq_argmin(
    const __bf16* __restrict__ zh, const __bf16* __restrict__ zl,
    const __bf16* __restrict__ eh, const __bf16* __restrict__ el,
    const float* __restrict__ en2,
    float* __restrict__ pbv, int* __restrict__ pbi) {
  __shared__ __align__(16) unsigned char smem[2 * BUFB];

  const int tid  = threadIdx.x;
  const int lane = tid & 31;
  const int wave = tid >> 5;
  const int col  = lane & 15;
  const int half = lane >> 4;
  const int t0   = (blockIdx.x * 8 + wave) * 32;
  const int cbase = blockIdx.y * (NT * 16);

  // ---- resident A fragments: 2 token strips x (hi,lo) x 8 channel chunks.
  // A 16x32 bf16 layout: lane row = lane&15; K = c*32 + {0..7,16..23} (half 0)
  //                                          K = c*32 + {8..15,24..31} (half 1)
  AFrag a0h[8], a0l[8], a1h[8], a1l[8];
  {
    const __bf16* r0h = zh + (size_t)(t0 + col) * CH;
    const __bf16* r0l = zl + (size_t)(t0 + col) * CH;
    const __bf16* r1h = zh + (size_t)(t0 + 16 + col) * CH;
    const __bf16* r1l = zl + (size_t)(t0 + 16 + col) * CH;
#pragma unroll
    for (int c = 0; c < 8; ++c) {
      const int k0 = c * 32 + half * 8;
      a0h[c].q[0] = *(const uint4*)(r0h + k0);
      a0h[c].q[1] = *(const uint4*)(r0h + k0 + 16);
      a0l[c].q[0] = *(const uint4*)(r0l + k0);
      a0l[c].q[1] = *(const uint4*)(r0l + k0 + 16);
      a1h[c].q[0] = *(const uint4*)(r1h + k0);
      a1h[c].q[1] = *(const uint4*)(r1h + k0 + 16);
      a1l[c].q[0] = *(const uint4*)(r1l + k0);
      a1l[c].q[1] = *(const uint4*)(r1l + k0 + 16);
    }
  }

  // ---- cooperative LDS staging: thread -> (code row, 32B chunk of row)
  const int srow  = tid >> 4;
  const int spart = tid & 15;
  const __bf16* g_h = eh + (size_t)(cbase + srow) * CH + spart * 16;
  const __bf16* g_l = el + (size_t)(cbase + srow) * CH + spart * 16;
  unsigned char* s_h = smem + srow * ROWB + spart * 32;
  unsigned char* s_l = smem + MATB + srow * ROWB + spart * 32;

  const float* en2t = en2 + cbase;

  uint4 rh0, rh1, rl0, rl1;
  rh0 = *(const uint4*)(g_h);     rh1 = *(const uint4*)(g_h + 8);
  rl0 = *(const uint4*)(g_l);     rl1 = *(const uint4*)(g_l + 8);
  float en_cur = en2t[col];       // ||e||^2 for tile 0 (this lane's column)
  *(uint4*)(s_h) = rh0;           *(uint4*)(s_h + 16) = rh1;
  *(uint4*)(s_l) = rl0;           *(uint4*)(s_l + 16) = rl1;
  __syncthreads();

  float bv0[8], bv1[8];
  int   bi0[8], bi1[8];
#pragma unroll
  for (int j = 0; j < 8; ++j) {
    bv0[j] = 3.4e38f; bv1[j] = 3.4e38f; bi0[j] = 0; bi1[j] = 0;
  }

  for (int nt = 0; nt < NT; ++nt) {
    const int buf = nt & 1;

    // Prefetch next tile (and its ||e||^2 lane value) -> hidden under WMMAs.
    float en_nxt = 0.0f;
    if (nt + 1 < NT) {
      const size_t go = (size_t)(nt + 1) * 16 * CH;
      rh0 = *(const uint4*)(g_h + go);  rh1 = *(const uint4*)(g_h + go + 8);
      rl0 = *(const uint4*)(g_l + go);  rl1 = *(const uint4*)(g_l + go + 8);
      en_nxt = en2t[(nt + 1) * 16 + col];
    }

    const float en = en_cur;
    v8f acc0 = {en, en, en, en, en, en, en, en};
    v8f acc1 = acc0;

    // B 32x16 bf16 layout: lane col = lane&15; K = chunk*32 + half*8 + {0..7,16..23}
    const unsigned char* fb = smem + buf * BUFB + col * ROWB + half * 16;
#pragma unroll
    for (int c = 0; c < 8; ++c) {
      AFrag bh, bl;
      bh.q[0] = *(const uint4*)(fb + c * 64);
      bh.q[1] = *(const uint4*)(fb + c * 64 + 32);
      bl.q[0] = *(const uint4*)(fb + MATB + c * 64);
      bl.q[1] = *(const uint4*)(fb + MATB + c * 64 + 32);

      acc0 = WMMA_BF16(a0h[c].v, bh.v, acc0);
      acc0 = WMMA_BF16(a0h[c].v, bl.v, acc0);
      acc0 = WMMA_BF16(a0l[c].v, bh.v, acc0);

      acc1 = WMMA_BF16(a1h[c].v, bh.v, acc1);
      acc1 = WMMA_BF16(a1h[c].v, bl.v, acc1);
      acc1 = WMMA_BF16(a1l[c].v, bh.v, acc1);
    }

    const int gi = cbase + nt * 16 + col;
#pragma unroll
    for (int j = 0; j < 8; ++j) {
      float v0 = acc0[j];
      if (v0 < bv0[j]) { bv0[j] = v0; bi0[j] = gi; }
      float v1 = acc1[j];
      if (v1 < bv1[j]) { bv1[j] = v1; bi1[j] = gi; }
    }

    // Publish next tile to the other LDS buffer.
    if (nt + 1 < NT) {
      const int off = ((nt + 1) & 1) * BUFB;
      *(uint4*)(s_h + off) = rh0;  *(uint4*)(s_h + off + 16) = rh1;
      *(uint4*)(s_l + off) = rl0;  *(uint4*)(s_l + off + 16) = rl1;
    }
    en_cur = en_nxt;
    __syncthreads();
  }

  // ---- per-token reduction across the 16 code-columns (16 lanes).
  const int yoff = blockIdx.y * TOKENS;
#pragma unroll
  for (int j = 0; j < 8; ++j) {
    {
      float v = bv0[j];
      int   i = bi0[j];
#pragma unroll
      for (int m = 1; m < 16; m <<= 1) {
        float ov = __shfl_xor(v, m, 32);
        int   oi = __shfl_xor(i, m, 32);
        if (ov < v || (ov == v && oi < i)) { v = ov; i = oi; }
      }
      if (col == 0) {
        pbv[yoff + t0 + j + half * 8] = v;
        pbi[yoff + t0 + j + half * 8] = i;
      }
    }
    {
      float v = bv1[j];
      int   i = bi1[j];
#pragma unroll
      for (int m = 1; m < 16; m <<= 1) {
        float ov = __shfl_xor(v, m, 32);
        int   oi = __shfl_xor(i, m, 32);
        if (ov < v || (ov == v && oi < i)) { v = ov; i = oi; }
      }
      if (col == 0) {
        pbv[yoff + t0 + 16 + j + half * 8] = v;
        pbi[yoff + t0 + 16 + j + half * 8] = i;
      }
    }
  }
}

// ---------------------------------------------------------------------------
// Merge the two code-half argmins (half 0 wins ties -> first occurrence).
// ---------------------------------------------------------------------------
__global__ void combine(const float* __restrict__ pbv,
                        const int* __restrict__ pbi,
                        int* __restrict__ idx) {
  const int n = blockIdx.x * 256 + threadIdx.x;
  float v0 = pbv[n], v1 = pbv[TOKENS + n];
  int   i0 = pbi[n], i1 = pbi[TOKENS + n];
  idx[n] = (v1 < v0) ? i1 : i0;
}

// ---------------------------------------------------------------------------
// Gather z_q (original f32 embedding) into NCHW output, emit indices as
// floats, and per-token squared-error partials (deterministic reduction).
// ---------------------------------------------------------------------------
__global__ void gather_out(const float* __restrict__ z,
                           const float* __restrict__ emb,
                           const int* __restrict__ idx,
                           float* __restrict__ out,
                           float* __restrict__ part) {
  __shared__ float red[256];
  const int n = blockIdx.x;
  const int c = threadIdx.x;
  const int b = n >> 10;
  const int hw = n & 1023;
  const int k = idx[n];
  float e = emb[(size_t)k * CH + c];
  size_t zoff = ((size_t)(b * CH + c) << 10) + hw;
  float zv = z[zoff];
  out[zoff] = e;  // straight-through forward value == z_q
  float d = e - zv;
  red[c] = d * d;
  __syncthreads();
  for (int s = 128; s > 0; s >>= 1) {
    if (c < s) red[c] += red[c + s];
    __syncthreads();
  }
  if (c == 0) part[n] = red[0];
  if (c == 1) out[(size_t)NC + 3 + n] = (float)k;  // indices output
}

// ---------------------------------------------------------------------------
// Fixed-order final reduction -> the three scalar losses.
// ---------------------------------------------------------------------------
__global__ void finalize(const float* __restrict__ part,
                         float* __restrict__ out) {
  __shared__ float red[256];
  const int t = threadIdx.x;
  float s = 0.0f;
  for (int i = t; i < TOKENS; i += 256) s += part[i];
  red[t] = s;
  __syncthreads();
  for (int w = 128; w > 0; w >>= 1) {
    if (t < w) red[t] += red[t + w];
    __syncthreads();
  }
  if (t == 0) {
    float mse = red[0] / (float)NC;
    out[NC + 0] = 1.25f * mse;  // loss
    out[NC + 1] = 0.25f * mse;  // commitment_loss
    out[NC + 2] = mse;          // codebook_loss
  }
}

extern "C" void kernel_launch(void* const* d_in, const int* in_sizes, int n_in,
                              void* d_out, int out_size, void* d_ws, size_t ws_size,
                              hipStream_t stream) {
  const float* z   = (const float*)d_in[0];   // [32,256,32,32] f32
  const float* emb = (const float*)d_in[1];   // [8192,256] f32
  float* out = (float*)d_out;

  char* ws = (char*)d_ws;
  __bf16* zh  = (__bf16*)(ws + 0);                    // 16 MB
  __bf16* zl  = (__bf16*)(ws + (size_t)16777216);     // 16 MB
  __bf16* eh  = (__bf16*)(ws + (size_t)33554432);     //  4 MB
  __bf16* el  = (__bf16*)(ws + (size_t)37748736);     //  4 MB
  float*  en2 = (float*) (ws + (size_t)41943040);     // 32 KB
  int*    idx = (int*)   (ws + (size_t)41975808);     // 128 KB
  float*  prt = (float*) (ws + (size_t)42106880);     // 128 KB
  float*  pbv = (float*) (ws + (size_t)42237952);     // 256 KB
  int*    pbi = (int*)   (ws + (size_t)42500096);     // 256 KB

  prep_z   <<<TOKENS, 256, 0, stream>>>(z, zh, zl);
  prep_e   <<<CODES,  256, 0, stream>>>(emb, eh, el, en2);
  vq_argmin<<<dim3(TOKENS / 256, 2), 256, 0, stream>>>(zh, zl, eh, el, en2, pbv, pbi);
  combine  <<<TOKENS / 256, 256, 0, stream>>>(pbv, pbi, idx);
  gather_out<<<TOKENS, 256, 0, stream>>>(z, emb, idx, out, prt);
  finalize <<<1, 256, 0, stream>>>(prt, out);
}
